// FullyHyperbolicBlock_9964324127134
// MI455X (gfx1250) — matrix operand order repaired and checked by reading, not compiled, not run on hardware
//
#include <hip/hip_runtime.h>
#include <hip/hip_bf16.h>
#include <math.h>

typedef __attribute__((ext_vector_type(16))) __bf16 v16bf;
typedef __attribute__((ext_vector_type(8)))  __bf16 v8bf;
typedef __attribute__((ext_vector_type(8)))  float  v8f;

#define B_   4
#define T_   1024
#define NE_  768
#define NH_  12
#define HD_  64
#define ROWS (B_ * T_)   // 4096
#define EPS_ 1e-6f
#define NEG_BIG -1e30f

// ---------------------------------------------------------------------------
// WMMA helpers (CDNA5 wave32, 16x16x32 bf16 -> f32)
// ---------------------------------------------------------------------------
__device__ inline v8f wmma_bf16(v16bf a, v16bf b, v8f c) {
  return __builtin_amdgcn_wmma_f32_16x16x32_bf16(false, a, false, b, (short)0, c, false, false);
}

__device__ inline v16bf join8(v8bf lo, v8bf hi) {
  return __builtin_shufflevector(lo, hi, 0,1,2,3,4,5,6,7,8,9,10,11,12,13,14,15);
}

// A-matrix fragment 16x32 bf16, row-major source [row, k], ld in elements.
// Per ISA: lanes 0-15 -> rows 0-15 hold K {0..7,16..23}; lanes 16-31 hold K {8..15,24..31}
__device__ inline v16bf ld_frag_a(const __bf16* A, int row0, int k0, int ld, int lane) {
  int m = lane & 15, hf = lane >> 4;
  const __bf16* p = A + (size_t)(row0 + m) * ld + k0 + 8 * hf;
  return join8(*(const v8bf*)p, *(const v8bf*)(p + 16));
}

// B-matrix fragment 32x16 bf16 from W[n, k] row-major (so B = W^T).
// Per ISA: lanes 0-15 -> cols, K {0..7,16..23}; lanes 16-31 -> K {8..15,24..31}.
__device__ inline v16bf ld_frag_b(const __bf16* W, int n0, int k0, int ld, int lane) {
  int n = lane & 15, hf = lane >> 4;
  const __bf16* p = W + (size_t)(n0 + n) * ld + k0 + 8 * hf;
  return join8(*(const v8bf*)p, *(const v8bf*)(p + 16));
}

// half-wave (16 lane) reductions; result replicated across the 16-lane group
__device__ inline float hred_sum(float v) {
  for (int o = 8; o; o >>= 1) v += __shfl_xor(v, o, 32);
  return v;
}
__device__ inline float hred_max(float v) {
  for (int o = 8; o; o >>= 1) v = fmaxf(v, __shfl_xor(v, o, 32));
  return v;
}

__device__ inline float block_sum(float v, float* sm) {
  int t = threadIdx.x;
  sm[t] = v; __syncthreads();
  for (int s = 128; s > 0; s >>= 1) {
    if (t < s) sm[t] += sm[t + s];
    __syncthreads();
  }
  float r = sm[0]; __syncthreads();
  return r;
}

// ---------------------------------------------------------------------------
// Weight convert + pad (f32 [N,K] -> bf16 [Npad,Kpad], zero fill)
// ---------------------------------------------------------------------------
__global__ void __launch_bounds__(256)
conv_pad_kernel(const float* __restrict__ W, __bf16* __restrict__ Wp,
                int N, int K, int Npad, int Kpad) {
  size_t idx = (size_t)blockIdx.x * 256 + threadIdx.x;
  size_t tot = (size_t)Npad * Kpad;
  if (idx >= tot) return;
  int n = (int)(idx / Kpad), k = (int)(idx % Kpad);
  Wp[idx] = (n < N && k < K) ? (__bf16)W[(size_t)n * K + k] : (__bf16)0.0f;
}

// ---------------------------------------------------------------------------
// GEMM: C[M=4096, Nout] = A[4096, Kpad]bf16 * W[Npad, Kpad]bf16^T (+bias)
// 128 threads = 4 waves in a 2x2 grid; each wave computes a 64x64 tile:
// 16 v_wmma per K=32 step against 16 b128 loads (1.0 loads/wmma).
// ---------------------------------------------------------------------------
__global__ void __launch_bounds__(128)
gemm_bf16_kernel(const __bf16* __restrict__ A, const __bf16* __restrict__ W,
                 const float* __restrict__ bias, float* __restrict__ C,
                 int Kpad, int ldc, int Nout) {
  const int lane = threadIdx.x & 31;
  const int wv   = threadIdx.x >> 5;
  const int m0   = blockIdx.x * 128 + (wv & 1) * 64;
  const int n0   = blockIdx.y * 128 + (wv >> 1) * 64;

  v8f acc[4][4];
  #pragma unroll
  for (int i = 0; i < 4; i++)
    #pragma unroll
    for (int j = 0; j < 4; j++)
      #pragma unroll
      for (int e = 0; e < 8; e++) acc[i][j][e] = 0.0f;

  for (int k = 0; k < Kpad; k += 32) {
    // prefetch the K-tile 8 steps ahead (k+256) (gfx1250 global_prefetch_b8); speculative, cheap
    if (k + 256 < Kpad) {
      __builtin_prefetch(A + (size_t)(m0 + (lane & 15)) * Kpad + k + 256, 0, 1);
      __builtin_prefetch(W + (size_t)(n0 + (lane & 15)) * Kpad + k + 256, 0, 1);
    }
    v16bf bfr[4], afr[4];
    #pragma unroll
    for (int j = 0; j < 4; j++) bfr[j] = ld_frag_b(W, n0 + 16 * j, k, Kpad, lane);
    #pragma unroll
    for (int i = 0; i < 4; i++) afr[i] = ld_frag_a(A, m0 + 16 * i, k, Kpad, lane);
    #pragma unroll
    for (int i = 0; i < 4; i++)
      #pragma unroll
      for (int j = 0; j < 4; j++)
        acc[i][j] = wmma_bf16(afr[i], bfr[j], acc[i][j]);
  }

  const int hf = lane >> 4, ln = lane & 15;
  #pragma unroll
  for (int i = 0; i < 4; i++)
    #pragma unroll
    for (int j = 0; j < 4; j++) {
      int col = n0 + 16 * j + ln;
      if (col < Nout) {
        float bb = bias ? bias[col] : 0.0f;
        #pragma unroll
        for (int s = 0; s < 8; s++) {
          int row = m0 + 16 * i + s + 8 * hf;
          C[(size_t)row * ldc + col] = acc[i][j][s] + bb;
        }
      }
    }
}

// ---------------------------------------------------------------------------
// Row-wise epilogue kernels (256 threads per 768/3075-wide row)
// ---------------------------------------------------------------------------
__global__ void __launch_bounds__(256)
project_in_kernel(const float* __restrict__ x, const float* __restrict__ cb,
                  float* __restrict__ lx) {
  __shared__ float sm[256];
  int row = blockIdx.x;
  const float* xr = x + (size_t)row * 768;
  float v[3]; float ss = 0.0f;
  #pragma unroll
  for (int i = 0; i < 3; i++) { v[i] = xr[threadIdx.x + 256 * i]; ss += v[i] * v[i]; }
  float tot = block_sum(ss, sm);
  float* orow = lx + (size_t)row * 769;
  #pragma unroll
  for (int i = 0; i < 3; i++) orow[1 + threadIdx.x + 256 * i] = v[i];
  if (threadIdx.x == 0) orow[0] = sqrtf(expf(cb[0]) + tot);
}

__global__ void __launch_bounds__(256)
skip_project_kernel(const float* __restrict__ lxin, const float* __restrict__ d,
                    const float* __restrict__ cb, float* __restrict__ out) {
  __shared__ float sm[256];
  int row = blockIdx.x;
  const float* ir = lxin + (size_t)row * 769 + 1;
  const float* dr = d + (size_t)row * 768;
  float v[3]; float ss = 0.0f;
  #pragma unroll
  for (int i = 0; i < 3; i++) {
    int idx = threadIdx.x + 256 * i;
    v[i] = ir[idx] + dr[idx]; ss += v[i] * v[i];
  }
  float tot = block_sum(ss, sm);
  float* orow = out + (size_t)row * 769;
  #pragma unroll
  for (int i = 0; i < 3; i++) orow[1 + threadIdx.x + 256 * i] = v[i];
  if (threadIdx.x == 0) orow[0] = sqrtf(expf(cb[0]) + tot);
}

// layernorm(lx[...,1:]) then project -> bf16 row [800]: col0=time, 1..768=y, pad=0
__global__ void __launch_bounds__(256)
ln_project_kernel(const float* __restrict__ lx, const float* __restrict__ g,
                  const float* __restrict__ bta, const float* __restrict__ cb,
                  __bf16* __restrict__ Aout) {
  __shared__ float sm[256];
  int row = blockIdx.x;
  const float* vr = lx + (size_t)row * 769 + 1;
  float v[3]; float s1 = 0.0f;
  #pragma unroll
  for (int i = 0; i < 3; i++) { v[i] = vr[threadIdx.x + 256 * i]; s1 += v[i]; }
  float mean = block_sum(s1, sm) * (1.0f / 768.0f);
  float s2 = 0.0f;
  #pragma unroll
  for (int i = 0; i < 3; i++) { float dd = v[i] - mean; s2 += dd * dd; }
  float var = block_sum(s2, sm) * (1.0f / 768.0f);
  float rstd = rsqrtf(var + 1e-5f);
  float y[3]; float ss = 0.0f;
  #pragma unroll
  for (int i = 0; i < 3; i++) {
    int idx = threadIdx.x + 256 * i;
    y[i] = (v[i] - mean) * rstd * g[idx] + bta[idx];
    ss += y[i] * y[i];
  }
  float tot = block_sum(ss, sm);
  __bf16* orow = Aout + (size_t)row * 800;
  #pragma unroll
  for (int i = 0; i < 3; i++) orow[1 + threadIdx.x + 256 * i] = (__bf16)y[i];
  if (threadIdx.x == 0) orow[0] = (__bf16)sqrtf(expf(cb[0]) + tot);
  if (threadIdx.x >= 1 && threadIdx.x <= 31) orow[768 + threadIdx.x] = (__bf16)0.0f;
}

// gelu(h[3075]) then project -> bf16 [3104]: col0=time, 1..3075=gelu, pad=0
__global__ void __launch_bounds__(256)
gelu_project_kernel(const float* __restrict__ h, const float* __restrict__ cm,
                    __bf16* __restrict__ gout) {
  __shared__ float sm[256];
  int row = blockIdx.x;
  const float* hr = h + (size_t)row * 3075;
  __bf16* orow = gout + (size_t)row * 3104;
  float ss = 0.0f;
  for (int i = threadIdx.x; i < 3075; i += 256) {
    float xv = hr[i];
    float ge = 0.5f * xv * (1.0f + erff(xv * 0.70710678f));
    ss += ge * ge;
    orow[1 + i] = (__bf16)ge;
  }
  float tot = block_sum(ss, sm);
  if (threadIdx.x == 0) orow[0] = (__bf16)sqrtf(expf(cm[0]) + tot);
  for (int i = 3076 + threadIdx.x; i < 3104; i += 256) orow[i] = (__bf16)0.0f;
}

// ---------------------------------------------------------------------------
// QKV preprocessing: rotary + Lorentz projection + V transpose.
// One wave per (b,t,h). Outputs: Qb/Kb [BH,T,64] bf16 rotated,
// Vt [BH,80,T] bf16 (rows 0..63 spatial, row 64 = v0 time, 65..79 = 0),
// q0/k0 [BH,T] f32 time components.
// ---------------------------------------------------------------------------
__global__ void __launch_bounds__(256)
qkv_rope_kernel(const float* __restrict__ qkv, const float* __restrict__ curv,
                __bf16* __restrict__ Qb, __bf16* __restrict__ Kb,
                __bf16* __restrict__ Vt, float* __restrict__ q0a,
                float* __restrict__ k0a) {
  int w = blockIdx.x * 8 + (threadIdx.x >> 5);
  int lane = threadIdx.x & 31;
  int h = w % NH_;
  int t = (w / NH_) % T_;
  int b = w / (NH_ * T_);
  const float* row = qkv + (size_t)(b * T_ + t) * (3 * NE_);
  float Kc = expf(curv[h]);

  float inv_freq = powf(10000.0f, -((float)(2 * lane) / 64.0f));
  float ang = (float)t * inv_freq;
  float c = cosf(ang), s = sinf(ang);

  float q1 = row[h * 64 + lane],         q2 = row[h * 64 + 32 + lane];
  float k1 = row[NE_ + h * 64 + lane],   k2 = row[NE_ + h * 64 + 32 + lane];
  float v1 = row[2*NE_ + h * 64 + lane], v2 = row[2*NE_ + h * 64 + 32 + lane];
  float qr1 =  q1 * c + q2 * s, qr2 = -q1 * s + q2 * c;
  float kr1 =  k1 * c + k2 * s, kr2 = -k1 * s + k2 * c;

  float nq = qr1 * qr1 + qr2 * qr2;
  float nk = kr1 * kr1 + kr2 * kr2;
  float nv = v1 * v1 + v2 * v2;
  for (int o = 16; o; o >>= 1) {
    nq += __shfl_xor(nq, o, 32);
    nk += __shfl_xor(nk, o, 32);
    nv += __shfl_xor(nv, o, 32);
  }

  size_t bh = (size_t)(b * NH_ + h);
  size_t base = (bh * T_ + t) * 64;
  Qb[base + lane]      = (__bf16)qr1;
  Qb[base + 32 + lane] = (__bf16)qr2;
  Kb[base + lane]      = (__bf16)kr1;
  Kb[base + 32 + lane] = (__bf16)kr2;

  size_t vbase = bh * 80 * T_ + t;
  Vt[vbase + (size_t)lane * T_]        = (__bf16)v1;
  Vt[vbase + (size_t)(lane + 32) * T_] = (__bf16)v2;
  if (lane == 0) {
    Vt[vbase + (size_t)64 * T_] = (__bf16)sqrtf(Kc + nv);
    q0a[bh * T_ + t] = sqrtf(Kc + nq);
    k0a[bh * T_ + t] = sqrtf(Kc + nk);
  }
  if (lane >= 1 && lane <= 15)
    Vt[vbase + (size_t)(64 + lane) * T_] = (__bf16)0.0f;
}

// ---------------------------------------------------------------------------
// Flash-style Lorentz attention. Block = 4 waves, each wave 16 query rows.
// Per j-tile(32): 4 WMMA for S = Q.K^T, elementwise distance+online softmax,
// P transposed via per-wave LDS (s_wait_dscnt), 5 WMMA for acc += P.Vt.
// Output: aout [4096, 800] bf16, per-head 65-block = [time, spatial x 64].
// ---------------------------------------------------------------------------
__global__ void __launch_bounds__(128)
attn_kernel(const __bf16* __restrict__ Qb, const __bf16* __restrict__ Kb,
            const __bf16* __restrict__ Vt, const float* __restrict__ q0a,
            const float* __restrict__ k0a, const float* __restrict__ curv,
            __bf16* __restrict__ aout) {
  __shared__ __bf16 pshm[4][16 * 40];
  const int lane = threadIdx.x & 31;
  const int wv   = threadIdx.x >> 5;
  const int hf   = lane >> 4;
  const int ln   = lane & 15;

  const int ib = blockIdx.x & 15;
  const int h  = (blockIdx.x >> 4) % NH_;
  const int b  = blockIdx.x / (16 * NH_);
  const int i0 = ib * 64 + wv * 16;

  const size_t bh = (size_t)(b * NH_ + h);
  const __bf16* Qh = Qb + bh * T_ * 64;
  const __bf16* Kh = Kb + bh * T_ * 64;
  const __bf16* Vh = Vt + bh * 80 * T_;
  const float* q0h = q0a + bh * T_;
  const float* k0h = k0a + bh * T_;

  const float Kc = expf(curv[h]);
  const float sK = sqrtf(Kc);
  const float rK = 1.0f / Kc;

  v16bf qf0 = ld_frag_a(Qh, i0, 0, 64, lane);
  v16bf qf1 = ld_frag_a(Qh, i0, 32, 64, lane);
  float q0r[8];
  #pragma unroll
  for (int s = 0; s < 8; s++) q0r[s] = q0h[i0 + s + 8 * hf];

  v8f acc[5];
  #pragma unroll
  for (int f = 0; f < 5; f++)
    #pragma unroll
    for (int e = 0; e < 8; e++) acc[f][e] = 0.0f;
  float mrun[8], lrun[8];
  #pragma unroll
  for (int s = 0; s < 8; s++) { mrun[s] = NEG_BIG; lrun[s] = 0.0f; }

  const int jtmax = (i0 + 15) >> 5;
  for (int jt = 0; jt <= jtmax; jt++) {
    const int j0 = jt * 32;
    // S = Q * K^T (16 x 32), accumulate over d = 64 in two K=32 steps
    v8f sf[2];
    #pragma unroll
    for (int fn = 0; fn < 2; fn++) {
      v16bf b0 = ld_frag_b(Kh, j0 + 16 * fn, 0, 64, lane);
      v16bf b1 = ld_frag_b(Kh, j0 + 16 * fn, 32, 64, lane);
      v8f z;
      #pragma unroll
      for (int e = 0; e < 8; e++) z[e] = 0.0f;
      z = wmma_bf16(qf0, b0, z);
      sf[fn] = wmma_bf16(qf1, b1, z);
    }

    const int jA = j0 + ln, jB = j0 + 16 + ln;
    const float k0A = k0h[jA], k0B = k0h[jB];

    float p[2][8], scl[8];
    #pragma unroll
    for (int s = 0; s < 8; s++) {
      const int i = i0 + s + 8 * hf;
      float sc0, sc1;
      {
        float inner = sf[0][s] - q0r[s] * k0A;
        if (jA > i) sc0 = NEG_BIG;
        else {
          float cc = fmaxf(-inner * rK, 1.0f + EPS_);
          sc0 = -sK * logf(cc + sqrtf(cc * cc - 1.0f));
        }
      }
      {
        float inner = sf[1][s] - q0r[s] * k0B;
        if (jB > i) sc1 = NEG_BIG;
        else {
          float cc = fmaxf(-inner * rK, 1.0f + EPS_);
          sc1 = -sK * logf(cc + sqrtf(cc * cc - 1.0f));
        }
      }
      float rowm = hred_max(fmaxf(sc0, sc1));
      float mnew = fmaxf(mrun[s], rowm);
      scl[s] = expf(mrun[s] - mnew);
      float p0 = expf(sc0 - mnew), p1 = expf(sc1 - mnew);
      p[0][s] = p0; p[1][s] = p1;
      float rs = hred_sum(p0 + p1);
      lrun[s] = lrun[s] * scl[s] + rs;
      mrun[s] = mnew;
    }

    #pragma unroll
    for (int f = 0; f < 5; f++)
      #pragma unroll
      for (int s = 0; s < 8; s++) acc[f][s] *= scl[s];

    // P (D-layout f32) -> LDS -> A-layout bf16 fragment
    __bf16* pb = &pshm[wv][0];
    #pragma unroll
    for (int fn = 0; fn < 2; fn++)
      #pragma unroll
      for (int s = 0; s < 8; s++)
        pb[(s + 8 * hf) * 40 + 16 * fn + ln] = (__bf16)p[fn][s];
    asm volatile("s_wait_dscnt 0" ::: "memory");
    v16bf pf = ld_frag_a(pb, 0, 0, 40, lane);

    #pragma unroll
    for (int f = 0; f < 5; f++) {
      v16bf vf = ld_frag_b(Vh, 16 * f, j0, T_, lane);
      acc[f] = wmma_bf16(pf, vf, acc[f]);
    }
  }

  // finalize: softmax normalize, Lorentz normalize, store
  float il[8];
  #pragma unroll
  for (int s = 0; s < 8; s++) il[s] = 1.0f / lrun[s];
  #pragma unroll
  for (int f = 0; f < 5; f++)
    #pragma unroll
    for (int s = 0; s < 8; s++) acc[f][s] *= il[s];

  float nrm[8];
  #pragma unroll
  for (int s = 0; s < 8; s++) {
    float part = 0.0f;
    #pragma unroll
    for (int f = 0; f < 4; f++) part += acc[f][s] * acc[f][s];
    part = hred_sum(part);                               // ||a_spatial||^2
    float a0 = hred_sum((ln == 0) ? acc[4][s] : 0.0f);   // broadcast time comp
    nrm[s] = sK * rsqrtf(fmaxf(a0 * a0 - part, EPS_));
  }

  const size_t colbase = (size_t)h * 65;
  #pragma unroll
  for (int f = 0; f < 4; f++)
    #pragma unroll
    for (int s = 0; s < 8; s++) {
      int trow = i0 + s + 8 * hf;
      aout[((size_t)(b * T_ + trow)) * 800 + colbase + 1 + 16 * f + ln] =
          (__bf16)(acc[f][s] * nrm[s]);
    }
  #pragma unroll
  for (int s = 0; s < 8; s++) {
    if (ln == 0) {
      int trow = i0 + s + 8 * hf;
      aout[((size_t)(b * T_ + trow)) * 800 + colbase] = (__bf16)(acc[4][s] * nrm[s]);
    }
  }
}

// ---------------------------------------------------------------------------
// Host launch
// ---------------------------------------------------------------------------
extern "C" void kernel_launch(void* const* d_in, const int* in_sizes, int n_in,
                              void* d_out, int out_size, void* d_ws, size_t ws_size,
                              hipStream_t stream) {
  (void)in_sizes; (void)n_in; (void)out_size; (void)ws_size;

  const float* x      = (const float*)d_in[0];
  const float* cb     = (const float*)d_in[1];
  const float* cm     = (const float*)d_in[2];
  const float* ch     = (const float*)d_in[3];
  const float* qkv_w  = (const float*)d_in[4];
  const float* proj_w = (const float*)d_in[5];
  const float* proj_b = (const float*)d_in[6];
  const float* exp_w  = (const float*)d_in[7];
  const float* exp_b  = (const float*)d_in[8];
  const float* shr_w  = (const float*)d_in[9];
  const float* shr_b  = (const float*)d_in[10];
  const float* ln_g   = (const float*)d_in[11];
  const float* ln_bt  = (const float*)d_in[12];
  float* out = (float*)d_out;

  char* ws = (char*)d_ws;
  size_t off = 0;
  auto alloc = [&](size_t bytes) -> char* {
    off = (off + 255) & ~(size_t)255;
    char* p = ws + off; off += bytes; return p;
  };

  float*  lx1   = (float*) alloc((size_t)ROWS * 769 * 4);
  float*  lx2   = (float*) alloc((size_t)ROWS * 769 * 4);
  __bf16* Abuf  = (__bf16*)alloc((size_t)ROWS * 800 * 2);
  __bf16* abuf  = (__bf16*)alloc((size_t)ROWS * 800 * 2);
  float*  attno = (float*) alloc((size_t)ROWS * 768 * 4);
  float*  mlpo  = (float*) alloc((size_t)ROWS * 768 * 4);
  __bf16* gbuf  = (__bf16*)alloc((size_t)ROWS * 3104 * 2);
  __bf16* Wqkv  = (__bf16*)alloc((size_t)2304 * 800 * 2);
  __bf16* Wproj = (__bf16*)alloc((size_t)768 * 800 * 2);
  __bf16* Wexp  = (__bf16*)alloc((size_t)3200 * 800 * 2);
  __bf16* Wshr  = (__bf16*)alloc((size_t)768 * 3104 * 2);

  // attention temporaries; region reused for MLP hidden (dead by then)
  off = (off + 255) & ~(size_t)255;
  size_t regoff = off;
  float*  qkv = (float*) alloc((size_t)ROWS * 2304 * 4);
  __bf16* Qb  = (__bf16*)alloc((size_t)B_ * NH_ * T_ * 64 * 2);
  __bf16* Kb  = (__bf16*)alloc((size_t)B_ * NH_ * T_ * 64 * 2);
  __bf16* Vt  = (__bf16*)alloc((size_t)B_ * NH_ * 80 * T_ * 2);
  float*  q0  = (float*) alloc((size_t)B_ * NH_ * T_ * 4);
  float*  k0  = (float*) alloc((size_t)B_ * NH_ * T_ * 4);
  float*  hbuf = (float*)(ws + regoff);   // aliases qkv..k0 (50.4MB <= 58.6MB)

  auto cvt = [&](const float* W, __bf16* Wp, int N, int K, int Npad, int Kpad) {
    size_t tot = (size_t)Npad * Kpad;
    conv_pad_kernel<<<(unsigned)((tot + 255) / 256), 256, 0, stream>>>(W, Wp, N, K, Npad, Kpad);
  };
  cvt(qkv_w,  Wqkv,  2304, 769, 2304, 800);
  cvt(proj_w, Wproj, 768,  780, 768,  800);
  cvt(exp_w,  Wexp,  3075, 769, 3200, 800);
  cvt(shr_w,  Wshr,  768, 3076, 768, 3104);

  // lx = project(x)
  project_in_kernel<<<ROWS, 256, 0, stream>>>(x, cb, lx1);

  // ---- attention branch ----
  ln_project_kernel<<<ROWS, 256, 0, stream>>>(lx1, ln_g, ln_bt, cb, Abuf);
  gemm_bf16_kernel<<<dim3(32, 18), 128, 0, stream>>>(Abuf, Wqkv, nullptr, qkv, 800, 2304, 2304);
  qkv_rope_kernel<<<(B_ * NH_ * T_) / 8, 256, 0, stream>>>(qkv, ch, Qb, Kb, Vt, q0, k0);
  hipMemsetAsync(abuf, 0, (size_t)ROWS * 800 * 2, stream);
  attn_kernel<<<B_ * NH_ * 16, 128, 0, stream>>>(Qb, Kb, Vt, q0, k0, ch, abuf);
  gemm_bf16_kernel<<<dim3(32, 6), 128, 0, stream>>>(abuf, Wproj, proj_b, attno, 800, 768, 768);
  skip_project_kernel<<<ROWS, 256, 0, stream>>>(lx1, attno, cb, lx2);

  // ---- MLP branch ----
  ln_project_kernel<<<ROWS, 256, 0, stream>>>(lx2, ln_g, ln_bt, cb, Abuf);
  gemm_bf16_kernel<<<dim3(32, 25), 128, 0, stream>>>(Abuf, Wexp, exp_b, hbuf, 800, 3075, 3075);
  gelu_project_kernel<<<ROWS, 256, 0, stream>>>(hbuf, cm, gbuf);
  gemm_bf16_kernel<<<dim3(32, 6), 128, 0, stream>>>(gbuf, Wshr, shr_b, mlpo, 3104, 768, 768);
  skip_project_kernel<<<ROWS, 256, 0, stream>>>(lx2, mlpo, cb, out);
}
